// MambaFiLMDecoder_19516331393479
// MI455X (gfx1250) — matrix operand order repaired and checked by reading, not compiled, not run on hardware
//
#include <hip/hip_runtime.h>
#include <hip/hip_bf16.h>
#include <math.h>

// ---------------- problem constants ----------------
#define DD   768
#define NL   6
#define DI   1536
#define RR   48
#define NS   16
#define KC   4
#define BB   2
#define LL   1024
#define SIXD (6*DD)
#define KPAD 64          // padded K for the w_dt GEMM (48 -> 64)

// ---------------- WMMA types (gfx1250, wave32) ----------------
typedef __bf16 bf16_t;
typedef bf16_t v16bf __attribute__((ext_vector_type(16)));
typedef float  v8f   __attribute__((ext_vector_type(8)));
typedef unsigned short ushort8 __attribute__((ext_vector_type(8)));

union BFrag { v16bf v; ushort8 h[2]; unsigned short u[16]; };

__device__ __forceinline__ unsigned short f2bf(float f) {
    union { float f; unsigned int u; } x; x.f = f;
    unsigned int u = x.u;
    unsigned int r = u + 0x7FFFu + ((u >> 16) & 1u);   // round-to-nearest-even
    return (unsigned short)(r >> 16);
}

__device__ __forceinline__ float silu_f(float x) {
    return x / (1.f + __expf(-x));
}

// =====================================================================
// GEMM: C(MxN, f32) = A(MxK, bf16, stride lda) * W(NxK, bf16, row-major)^T
// v_wmma_f32_16x16x32_bf16, 32x32 tile per wave (2 A-frags x 2 B-frags).
// Requires Kdim % 32 == 0, M % 32 == 0 (pad inputs instead of guarding).
// epilogue: 0 = none, 1 = softplus(v + aux[col]), 2 = v * aux[(row>>rowShift)*auxStride + col]
// =====================================================================
__global__ void __launch_bounds__(256)
wmma_gemm_bf16(const unsigned short* __restrict__ A, int lda,
               const unsigned short* __restrict__ W,
               float* __restrict__ C, int ldc,
               int M, int N, int Kdim,
               int epilogue, const float* __restrict__ aux, int auxStride, int rowShift)
{
    const int wave  = threadIdx.x >> 5;
    const int lane  = threadIdx.x & 31;
    const int tileN = blockIdx.x * 8 + wave;      // 32-wide tiles
    const int tileM = blockIdx.y;                 // 32-tall tiles
    const int col0  = tileN * 32;
    const int row0  = tileM * 32;
    if (col0 >= N || row0 >= M) return;           // wave-uniform exit
    const int nSub = (N - col0 >= 32) ? 2 : 1;    // wave-uniform ragged-N handling

    // A fragment (16x32 bf16): lane half selects K-offset 0/8; K = khA..khA+7 then 16+khA..16+khA+7 per lane.
    const int mA  = lane & 15;
    const int khA = (lane >> 4) << 3;             // 0 or 8
    // B fragment (32x16 bf16): lane half selects K-offset 0/8; same K order as A.
    const int nB  = lane & 15;
    const int khB = (lane >> 4) << 3;             // 0 or 8

    const unsigned short* Ar0 = A + (size_t)(row0 + mA) * lda;
    const unsigned short* Ar1 = Ar0 + (size_t)16 * lda;
    const unsigned short* Wr0 = W + (size_t)(col0 + nB) * Kdim;
    const unsigned short* Wr1 = Wr0 + (size_t)16 * Kdim;

    v8f acc00 = {0,0,0,0,0,0,0,0}, acc01 = acc00, acc10 = acc00, acc11 = acc00;

    for (int k0 = 0; k0 < Kdim; k0 += 32) {
        __builtin_prefetch(Wr0 + k0 + 64, 0, 1);  // weights stream from HBM/L2
        BFrag a0, a1, b0;
        a0.h[0] = *(const ushort8*)(Ar0 + k0 + khA);
        a0.h[1] = *(const ushort8*)(Ar0 + k0 + 16 + khA);
        a1.h[0] = *(const ushort8*)(Ar1 + k0 + khA);
        a1.h[1] = *(const ushort8*)(Ar1 + k0 + 16 + khA);
        b0.h[0] = *(const ushort8*)(Wr0 + k0 + khB);
        b0.h[1] = *(const ushort8*)(Wr0 + k0 + 16 + khB);
        acc00 = __builtin_amdgcn_wmma_f32_16x16x32_bf16(false, a0.v, false, b0.v, (short)0, acc00, false, false);
        acc10 = __builtin_amdgcn_wmma_f32_16x16x32_bf16(false, a1.v, false, b0.v, (short)0, acc10, false, false);
        if (nSub == 2) {                          // wave-uniform
            BFrag b1;
            b1.h[0] = *(const ushort8*)(Wr1 + k0 + khB);
            b1.h[1] = *(const ushort8*)(Wr1 + k0 + 16 + khB);
            acc01 = __builtin_amdgcn_wmma_f32_16x16x32_bf16(false, a0.v, false, b1.v, (short)0, acc01, false, false);
            acc11 = __builtin_amdgcn_wmma_f32_16x16x32_bf16(false, a1.v, false, b1.v, (short)0, acc11, false, false);
        }
    }

    // C/D layout: VGPR r -> M = r + 8*(lane>=16), N = lane&15
    const int nOut  = lane & 15;
    const int mhalf = (lane >> 4) << 3;
    #pragma unroll
    for (int sub = 0; sub < 4; ++sub) {
        const int si = sub >> 1;                  // m sub-tile
        const int sj = sub & 1;                   // n sub-tile
        if (sj >= nSub) continue;                 // wave-uniform
        const v8f accv = (sub == 0) ? acc00 : (sub == 1) ? acc01 : (sub == 2) ? acc10 : acc11;
        #pragma unroll
        for (int r = 0; r < 8; ++r) {
            int row = row0 + 16 * si + mhalf + r;
            int col = col0 + 16 * sj + nOut;
            float v = accv[r];
            if (epilogue == 1) {
                v += aux[col];
                v = (v > 20.f) ? v : log1pf(__expf(v));         // softplus
            } else if (epilogue == 2) {
                v *= aux[(row >> rowShift) * auxStride + col];  // gate
            }
            C[(size_t)row * ldc + col] = v;
        }
    }
}

// =====================================================================
// f32 -> bf16 conversion (weights / activations)
// =====================================================================
__global__ void __launch_bounds__(256)
cvt_bf16_kernel(const float* __restrict__ in, unsigned short* __restrict__ out, int n)
{
    int i = blockIdx.x * 256 + threadIdx.x;
    if (i < n) out[i] = f2bf(in[i]);
}

// f32 (rows x kstride, first kvalid cols) -> bf16 (rows x kout, zero padded)
__global__ void __launch_bounds__(256)
cvt_bf16_pad_kernel(const float* __restrict__ in, unsigned short* __restrict__ out,
                    int rows, int kvalid, int kstride, int kout)
{
    int i = blockIdx.x * 256 + threadIdx.x;
    if (i >= rows * kout) return;
    int r = i / kout, k = i % kout;
    out[i] = (k < kvalid) ? f2bf(in[(size_t)r * kstride + k]) : (unsigned short)0;
}

// =====================================================================
// AdaLN modulation: mods[l][b][col] = silu(cond1[b]) . ada_w[l][col] + ada_b[l][col]
// =====================================================================
__global__ void __launch_bounds__(256)
mods_kernel(const float* __restrict__ cond1, const float* __restrict__ ada_w,
            const float* __restrict__ ada_b, float* __restrict__ mods)
{
    int idx = blockIdx.x * 256 + threadIdx.x;          // NL*BB*SIXD total
    int l   = idx / (BB * SIXD);
    int rem = idx % (BB * SIXD);
    int b   = rem / SIXD;
    int col = rem % SIXD;
    const float* wrow = ada_w + (size_t)l * SIXD * DD + (size_t)col * DD;
    const float* cv   = cond1 + b * DD;
    float acc = ada_b[l * SIXD + col];
    for (int dd = 0; dd < DD; ++dd)
        acc += silu_f(cv[dd]) * wrow[dd];
    mods[idx] = acc;
}

// =====================================================================
// residual add + LayerNorm + optional FiLM; optional bf16 output
// one 256-thread block per token row (D=768 -> 3 elems/thread)
// =====================================================================
__global__ void __launch_bounds__(256)
ln_film_kernel(const float* __restrict__ hin, const float* __restrict__ rin, int addRes,
               float* __restrict__ rout,
               const float* __restrict__ w, const float* __restrict__ bias,
               const float* __restrict__ modsL,   // null => no FiLM; layout [b][6*D]
               void* __restrict__ outp, int outBf)
{
    __shared__ float red[256];
    const int row = blockIdx.x;
    const int b   = row / LL;
    const int tid = threadIdx.x;
    float v[3];
    float s = 0.f;
    #pragma unroll
    for (int i = 0; i < 3; ++i) {
        int d = tid + i * 256;
        float x = hin[(size_t)row * DD + d];
        if (addRes) x += rin[(size_t)row * DD + d];
        if (rout) rout[(size_t)row * DD + d] = x;
        v[i] = x; s += x;
    }
    red[tid] = s; __syncthreads();
    for (int st = 128; st > 0; st >>= 1) { if (tid < st) red[tid] += red[tid + st]; __syncthreads(); }
    float mu = red[0] * (1.f / DD);
    __syncthreads();
    float s2 = 0.f;
    #pragma unroll
    for (int i = 0; i < 3; ++i) { float xc = v[i] - mu; s2 += xc * xc; }
    red[tid] = s2; __syncthreads();
    for (int st = 128; st > 0; st >>= 1) { if (tid < st) red[tid] += red[tid + st]; __syncthreads(); }
    float inv = rsqrtf(red[0] * (1.f / DD) + 1e-5f);
    #pragma unroll
    for (int i = 0; i < 3; ++i) {
        int d = tid + i * 256;
        float h = (v[i] - mu) * inv * w[d] + bias[d];
        if (modsL) {
            float shift = modsL[b * SIXD + d];
            float scale = modsL[b * SIXD + DD + d];
            h = shift + h * scale;
        }
        if (outBf) ((unsigned short*)outp)[(size_t)row * DD + d] = f2bf(h);
        else       ((float*)outp)[(size_t)row * DD + d] = h;
    }
}

// =====================================================================
// depthwise causal conv (K=4, left pad 3) + bias + SiLU on xi half of xz
// writes both f32 (for scan) and bf16 (for GEMM)
// =====================================================================
__global__ void __launch_bounds__(256)
conv_silu_kernel(const float* __restrict__ xz, const float* __restrict__ cw,
                 const float* __restrict__ cb, float* __restrict__ xi,
                 unsigned short* __restrict__ xi_bf)
{
    int idx = blockIdx.x * 256 + threadIdx.x;          // BB*LL*DI total
    int d = idx % DI;
    int t = (idx / DI) % LL;
    int b = idx / (DI * LL);
    float acc = cb[d];
    #pragma unroll
    for (int kk = 0; kk < KC; ++kk) {
        int tt = t - (KC - 1) + kk;
        if (tt >= 0)
            acc += cw[d * KC + kk] * xz[((size_t)(b * LL + tt)) * (2 * DI) + d];
    }
    float sv = silu_f(acc);
    xi[idx]    = sv;
    xi_bf[idx] = f2bf(sv);
}

// =====================================================================
// Selective scan: one lane per (b, d) channel, 16-state in registers,
// Bm/Cm staged through LDS in 64-step chunks. Fuses +x*d_skip and *silu(z).
// Output bf16 (feeds the w_out GEMM). grid = (DI/256, BB)
// =====================================================================
#define TCH 64
__global__ void __launch_bounds__(256)
scan_kernel(const float* __restrict__ dt, const float* __restrict__ xi,
            const float* __restrict__ xz, const float* __restrict__ dbc,
            const float* __restrict__ a_log, const float* __restrict__ dskip,
            unsigned short* __restrict__ y)
{
    __shared__ float sB[TCH * NS];
    __shared__ float sC[TCH * NS];
    const int d = blockIdx.x * 256 + threadIdx.x;
    const int b = blockIdx.y;
    float An[NS], h[NS];
    #pragma unroll
    for (int n = 0; n < NS; ++n) { An[n] = -__expf(a_log[d * NS + n]); h[n] = 0.f; }
    const float dsk = dskip[d];

    for (int tc = 0; tc < LL; tc += TCH) {
        for (int i = threadIdx.x; i < TCH * NS; i += 256) {
            int tt = i >> 4, n = i & 15;
            size_t base = (size_t)(b * LL + tc + tt) * (RR + 2 * NS);
            sB[i] = dbc[base + RR + n];
            sC[i] = dbc[base + RR + NS + n];
        }
        __syncthreads();
        for (int tt = 0; tt < TCH; ++tt) {
            size_t e = (size_t)(b * LL + tc + tt) * DI + d;
            float dtv = dt[e];
            float xv  = xi[e];
            float zv  = xz[(size_t)(b * LL + tc + tt) * (2 * DI) + DI + d];
            float dtx = dtv * xv;
            float acc = 0.f;
            #pragma unroll
            for (int n = 0; n < NS; ++n) {
                h[n] = __expf(dtv * An[n]) * h[n] + dtx * sB[tt * NS + n];
                acc += h[n] * sC[tt * NS + n];
            }
            y[e] = f2bf((acc + xv * dsk) * silu_f(zv));
        }
        __syncthreads();
    }
}

// =====================================================================
// host side
// =====================================================================
extern "C" void kernel_launch(void* const* d_in, const int* in_sizes, int n_in,
                              void* d_out, int out_size, void* d_ws, size_t ws_size,
                              hipStream_t stream) {
    (void)in_sizes; (void)n_in; (void)out_size; (void)ws_size;
    const float* x      = (const float*)d_in[0];
    const float* cond1  = (const float*)d_in[1];
    const float* ada_w  = (const float*)d_in[2];
    const float* ada_b  = (const float*)d_in[3];
    const float* norm_w = (const float*)d_in[4];
    const float* norm_b = (const float*)d_in[5];
    const float* out_w  = (const float*)d_in[6];
    const float* out_b  = (const float*)d_in[7];
    const float* w_in   = (const float*)d_in[8];
    const float* conv_w = (const float*)d_in[9];
    const float* conv_b = (const float*)d_in[10];
    const float* w_x    = (const float*)d_in[11];
    const float* w_dt   = (const float*)d_in[12];
    const float* b_dt   = (const float*)d_in[13];
    const float* a_log  = (const float*)d_in[14];
    const float* d_skip = (const float*)d_in[15];
    const float* w_out  = (const float*)d_in[16];
    float* out = (float*)d_out;

    const int ROWS = BB * LL;            // 2048

    // -------- workspace layout --------
    float* ws = (float*)d_ws;
    size_t off = 0;
    float* mods = ws + off; off += (size_t)NL * BB * SIXD;
    float* Rb   = ws + off; off += (size_t)ROWS * DD;          // residual
    float* xz   = ws + off; off += (size_t)ROWS * 2 * DI;      // in-proj (f32)
    float* xi   = ws + off; off += (size_t)ROWS * DI;          // conv+silu (f32, scan)
    float* dbc  = ws + off; off += (size_t)ROWS * (RR + 2*NS); // x-proj
    float* dtb  = ws + off; off += (size_t)ROWS * DI;          // softplus(dt)
    float* H    = ws + off; off += (size_t)ROWS * DD;          // hidden
    unsigned short* bfp = (unsigned short*)(ws + off);
    size_t boff = 0;
    unsigned short* xn_bf  = bfp + boff; boff += (size_t)ROWS * DD;     // LN+FiLM out
    unsigned short* xi_bf  = bfp + boff; boff += (size_t)ROWS * DI;     // conv out
    unsigned short* yb_bf  = bfp + boff; boff += (size_t)ROWS * DI;     // scan out
    unsigned short* dtc_bf = bfp + boff; boff += (size_t)ROWS * KPAD;   // dbc[:, :48] padded
    unsigned short* wb_in  = bfp + boff; boff += (size_t)2 * DI * DD;   // per-layer weights
    unsigned short* wb_x   = bfp + boff; boff += (size_t)(RR + 2*NS) * DI;
    unsigned short* wb_dt  = bfp + boff; boff += (size_t)DI * KPAD;     // padded 48 -> 64
    unsigned short* wb_out = bfp + boff; boff += (size_t)DD * DI;

    // AdaLN mods for all layers
    mods_kernel<<<(NL * BB * SIXD) / 256, 256, 0, stream>>>(cond1, ada_w, ada_b, mods);

    for (int l = 0; l < NL; ++l) {
        const float* hin = (l == 0) ? x : H;
        // residual += hidden; xn_bf = bf16(FiLM(LN(residual)))
        ln_film_kernel<<<ROWS, 256, 0, stream>>>(
            hin, Rb, (l > 0) ? 1 : 0, Rb,
            norm_w + (size_t)l * DD, norm_b + (size_t)l * DD,
            mods + (size_t)l * BB * SIXD, xn_bf, 1);

        // weight -> bf16 (per layer, reused buffers)
        cvt_bf16_kernel<<<(2 * DI * DD) / 256, 256, 0, stream>>>(
            w_in + (size_t)l * 2 * DI * DD, wb_in, 2 * DI * DD);

        // xz = xn @ w_in^T   (2048 x 3072, K=768): 96 N-tiles of 32
        wmma_gemm_bf16<<<dim3(12, ROWS / 32), 256, 0, stream>>>(
            xn_bf, DD, wb_in, xz, 2 * DI, ROWS, 2 * DI, DD, 0, nullptr, 0, 0);

        // depthwise conv + silu (f32 + bf16 outputs)
        conv_silu_kernel<<<(ROWS * DI) / 256, 256, 0, stream>>>(
            xz, conv_w + (size_t)l * DI * KC, conv_b + (size_t)l * DI, xi, xi_bf);

        cvt_bf16_kernel<<<((RR + 2*NS) * DI) / 256, 256, 0, stream>>>(
            w_x + (size_t)l * (RR + 2*NS) * DI, wb_x, (RR + 2*NS) * DI);

        // dbc = xi @ w_x^T   (2048 x 80, K=1536): 3 N-tiles of 32 (last ragged)
        wmma_gemm_bf16<<<dim3(1, ROWS / 32), 256, 0, stream>>>(
            xi_bf, DI, wb_x, dbc, RR + 2 * NS, ROWS, RR + 2 * NS, DI, 0, nullptr, 0, 0);

        // pad dbc[:, :48] and w_dt to K=64 bf16 (guard-free GEMM)
        cvt_bf16_pad_kernel<<<(ROWS * KPAD) / 256, 256, 0, stream>>>(
            dbc, dtc_bf, ROWS, RR, RR + 2 * NS, KPAD);
        cvt_bf16_pad_kernel<<<(DI * KPAD) / 256, 256, 0, stream>>>(
            w_dt + (size_t)l * DI * RR, wb_dt, DI, RR, RR, KPAD);

        // dt = softplus(dtc @ w_dt^T + b_dt)   (2048 x 1536, K=64)
        wmma_gemm_bf16<<<dim3(6, ROWS / 32), 256, 0, stream>>>(
            dtc_bf, KPAD, wb_dt, dtb, DI, ROWS, DI, KPAD,
            1, b_dt + (size_t)l * DI, 0, 0);

        // selective scan + d_skip + silu(z) gate -> bf16
        scan_kernel<<<dim3(DI / 256, BB), 256, 0, stream>>>(
            dtb, xi, xz, dbc,
            a_log + (size_t)l * DI * NS, d_skip + (size_t)l * DI, yb_bf);

        cvt_bf16_kernel<<<(DD * DI) / 256, 256, 0, stream>>>(
            w_out + (size_t)l * DD * DI, wb_out, DD * DI);

        // hidden = gate * (y @ w_out^T)   (2048 x 768, K=1536)
        wmma_gemm_bf16<<<dim3(3, ROWS / 32), 256, 0, stream>>>(
            yb_bf, DI, wb_out, H, DD, ROWS, DD, DI,
            2, mods + (size_t)l * BB * SIXD + 2 * DD, SIXD, 10 /* row>>10 = b */);
    }

    // final LayerNorm -> d_out (f32)
    ln_film_kernel<<<ROWS, 256, 0, stream>>>(
        H, nullptr, 0, nullptr, out_w, out_b, nullptr, out, 0);
}
